// VectorQuantizer_34892314313116
// MI455X (gfx1250) — compile-verified, matrix operand order checked by reading, not run on hardware
//
#include <hip/hip_runtime.h>
#include <hip/hip_bf16.h>
#include <math.h>

// ---------------- problem constants (from the reference) ----------------
#define NUM_EMB   4000
#define EMB_DIM   12544          // K, multiple of 32
#define N_TOK     2048           // M, multiple of 128
#define TM        128            // workgroup tile M (z rows)
#define TN        128            // workgroup tile N (embedding rows)
#define TK        32
#define NTHREADS  256            // 8 waves; wave tile = 32(M) x 64(N)

typedef __attribute__((ext_vector_type(16))) __bf16 v16bf;
typedef __attribute__((ext_vector_type(8)))  float  v8f;
typedef __attribute__((ext_vector_type(4)))  float  f32x4;
typedef __attribute__((ext_vector_type(4)))  __bf16 bf16x4;

union FragBF {
    v16bf v;
    int4  q[2];   // q[0] = VGPR0..3 (K 0..7 | 8..15), q[1] = VGPR4..7 (K 16..23 | 24..31)
};

__device__ __forceinline__ float block_reduce_sum_256(float v, float* smem) {
    int t = threadIdx.x;
    smem[t] = v;
    __syncthreads();
    for (int s = 128; s > 0; s >>= 1) {
        if (t < s) smem[t] += smem[t + s];
        __syncthreads();
    }
    float r = smem[0];
    __syncthreads();
    return r;
}

// ---------------- kernel 0: init argmin table ----------------
__global__ void vq_init_kernel(unsigned long long* __restrict__ amin) {
    int i = blockIdx.x * blockDim.x + threadIdx.x;
    if (i < N_TOK) amin[i] = 0xFFFFFFFFFFFFFFFFull;
}

// ---------------- kernel 1: normE[k] = sum(e_k^2) ----------------
__global__ __launch_bounds__(NTHREADS) void vq_norme_kernel(
        const float* __restrict__ emb, float* __restrict__ normE) {
    __shared__ float smem[NTHREADS];
    int k = blockIdx.x;
    const float* row = emb + (size_t)k * EMB_DIM;
    float acc = 0.f;
    for (int j = threadIdx.x; j < EMB_DIM; j += NTHREADS) {
        float v = row[j];
        acc = fmaf(v, v, acc);
    }
    float tot = block_reduce_sum_256(acc, smem);
    if (threadIdx.x == 0) normE[k] = tot;
}

// ---------------- kernel 2: bf16 WMMA scores + in-register argmin ----------------
// grid: (M/TM fastest, ceil(N/TN)): a 128-row B strip is reused across all M
// blocks while the whole A matrix (103 MB) stays hot in the 192 MB L2.
// Double-buffered LDS: one barrier per K step; next tile's global loads issue
// before the current tile's ds_load/WMMA stream so VMEM hides under math.
__global__ __launch_bounds__(NTHREADS) void vq_scores_argmin_kernel(
        const float* __restrict__ z, const float* __restrict__ emb,
        const float* __restrict__ normE,
        unsigned long long* __restrict__ amin) {
    __shared__ unsigned aT[2][TM * (TK / 2)];   // 2 x 8 KB
    __shared__ unsigned bT[2][TN * (TK / 2)];   // 2 x 8 KB
    __shared__ float    nE[TN];

    const int m0 = blockIdx.x * TM;
    const int n0 = blockIdx.y * TN;
    const int t    = threadIdx.x;
    const int lane = t & 31;
    const int wid  = t >> 5;
    const int mw   = wid & 3;                 // M wave coord -> rows mw*32
    const int nw   = wid >> 2;                // N wave coord -> cols nw*64

    if (t < TN) {
        int gn = n0 + t;
        nE[t] = (gn < NUM_EMB) ? normE[gn] : INFINITY;   // +inf never wins argmin
    }

    // staging coordinates (branchless: clamp OOB embedding rows, nE excludes them)
    const int srow = t >> 3;                  // 0..31 (+32 per step)
    const int sc4  = t & 7;                   // float4 slot in the 32-wide K strip
    const size_t zbase = (size_t)(m0 + srow) * EMB_DIM + sc4 * 4;

    // fragment addressing (16-bit A layout; B mirrored with lane = column)
    const int frow  = lane & 15;
    const int klo32 = (lane >> 4) << 2;       // lanes>=16 start at K=8 -> +4 u32
    const int aBase = (mw * 32) * 16;         // 16 u32 per LDS row
    const int bBase = (nw * 64) * 16;

    auto stage = [&](int kb, int buf) {
        #pragma unroll
        for (int i = 0; i < 4; ++i) {
            int row = srow + i * 32;          // 0..127
            f32x4 zv = *(const f32x4*)(z + zbase + (size_t)i * 32 * EMB_DIM + kb);
            *(bf16x4*)&aT[buf][row * 16 + sc4 * 2] = __builtin_convertvector(zv, bf16x4);
            int gn = n0 + row;
            if (gn > NUM_EMB - 1) gn = NUM_EMB - 1;     // clamp: always-legal load
            f32x4 ev = *(const f32x4*)(emb + (size_t)gn * EMB_DIM + kb + sc4 * 4);
            *(bf16x4*)&bT[buf][row * 16 + sc4 * 2] = __builtin_convertvector(ev, bf16x4);
        }
    };

    v8f c[2][4] = {};
    stage(0, 0);
    int buf = 0;
    for (int kb = 0; kb < EMB_DIM; kb += TK) {
        __syncthreads();                      // staging of `buf` visible; prev reads done
        const int cur = buf;
        if (kb + TK < EMB_DIM) {
            buf ^= 1;
            stage(kb + TK, buf);              // global loads overlap WMMAs below
        }

        FragBF a[2];
        #pragma unroll
        for (int i = 0; i < 2; ++i) {
            const unsigned* ap = &aT[cur][aBase + i * 256 + frow * 16 + klo32];
            a[i].q[0] = *(const int4*)(ap);
            a[i].q[1] = *(const int4*)(ap + 8);
        }
        #pragma unroll
        for (int f = 0; f < 4; ++f) {
            FragBF b;
            const unsigned* bp = &bT[cur][bBase + f * 256 + frow * 16 + klo32];
            b.q[0] = *(const int4*)(bp);
            b.q[1] = *(const int4*)(bp + 8);
            c[0][f] = __builtin_amdgcn_wmma_f32_16x16x32_bf16(false, a[0].v, false, b.v,
                                                              (short)0, c[0][f], false, false);
            c[1][f] = __builtin_amdgcn_wmma_f32_16x16x32_bf16(false, a[1].v, false, b.v,
                                                              (short)0, c[1][f], false, false);
        }
    }

    // C layout: lane<16 -> n=lane, m=r ; lane>=16 -> n=lane-16, m=8+r.
    // Per row: fold 4 N-frags in-lane, then shfl_xor min over the 16 lanes of the half.
    const int lhalf = (lane >> 4) << 3;       // 0 or 8
    #pragma unroll
    for (int i = 0; i < 2; ++i) {
        #pragma unroll
        for (int r = 0; r < 8; ++r) {
            float best = INFINITY;
            int   bidx = 0;
            #pragma unroll
            for (int f = 0; f < 4; ++f) {
                int nl = nw * 64 + f * 16 + (lane & 15);
                float d = nE[nl] - 2.0f * c[i][f][r];     // ||e||^2 - 2 z.e
                if (d < best) { best = d; bidx = n0 + nl; }
            }
            #pragma unroll
            for (int mask = 1; mask < 16; mask <<= 1) {
                float ok = __shfl_xor(best, mask, 32);
                int   oi = __shfl_xor(bidx, mask, 32);
                if (ok < best || (ok == best && oi < bidx)) { best = ok; bidx = oi; }
            }
            if ((lane & 15) == 0) {
                int m = m0 + mw * 32 + i * 16 + lhalf + r;
                unsigned ub = __float_as_uint(best);
                ub = (ub & 0x80000000u) ? ~ub : (ub | 0x80000000u);  // orderable key
                atomicMin(&amin[m], ((unsigned long long)ub << 32) | (unsigned)bidx);
            }
        }
    }
}

// ---------------- kernel 3: gather quantized + per-row loss partial ----------------
__global__ __launch_bounds__(NTHREADS) void vq_gather_kernel(
        const float* __restrict__ z, const float* __restrict__ emb,
        const unsigned long long* __restrict__ amin,
        float* __restrict__ out, float* __restrict__ partial) {
    __shared__ float smem[NTHREADS];
    const int m = blockIdx.x;
    int idx = (int)(amin[m] & 0xFFFFFFFFull);
    if (idx < 0) idx = 0;
    if (idx >= NUM_EMB) idx = NUM_EMB - 1;

    const float4* qr   = (const float4*)(emb + (size_t)idx * EMB_DIM);
    const float4* zr   = (const float4*)(z   + (size_t)m   * EMB_DIM);
    float4*       orow = (float4*)(out + (size_t)m * EMB_DIM);

    float acc = 0.f;
    for (int j = threadIdx.x; j < EMB_DIM / 4; j += NTHREADS) {
        float4 q  = qr[j];
        float4 zz = zr[j];
        orow[j] = q;                        // quantized_st forward value == quantized
        float dx = q.x - zz.x, dy = q.y - zz.y, dz = q.z - zz.z, dw = q.w - zz.w;
        acc = fmaf(dx, dx, acc); acc = fmaf(dy, dy, acc);
        acc = fmaf(dz, dz, acc); acc = fmaf(dw, dw, acc);
    }
    float tot = block_reduce_sum_256(acc, smem);
    if (threadIdx.x == 0) {
        partial[m] = tot;
        out[(size_t)N_TOK * EMB_DIM + 1 + m] = (float)idx;   // indices after loss slot
    }
}

// ---------------- kernel 4: final loss reduction ----------------
__global__ __launch_bounds__(NTHREADS) void vq_loss_kernel(
        const float* __restrict__ partial, float* __restrict__ out) {
    __shared__ float smem[NTHREADS];
    float acc = 0.f;
    for (int j = threadIdx.x; j < N_TOK; j += NTHREADS) acc += partial[j];
    float tot = block_reduce_sum_256(acc, smem);
    if (threadIdx.x == 0) {
        const double denom = (double)N_TOK * (double)EMB_DIM;
        // loss = q_latent + 0.25 * e_latent; both forward values equal the mse
        out[(size_t)N_TOK * EMB_DIM] = (float)(1.25 * ((double)tot / denom));
    }
}

extern "C" void kernel_launch(void* const* d_in, const int* in_sizes, int n_in,
                              void* d_out, int out_size, void* d_ws, size_t ws_size,
                              hipStream_t stream) {
    const float* z   = (const float*)d_in[0];
    const float* emb = (const float*)d_in[1];
    float* out = (float*)d_out;

    // workspace layout
    float*              ws_normE   = (float*)d_ws;                               // 4000 f32
    unsigned long long* ws_amin    = (unsigned long long*)((char*)d_ws + 16384); // 2048 u64
    float*              ws_partial = (float*)((char*)d_ws + 32768);              // 2048 f32

    vq_init_kernel<<<(N_TOK + NTHREADS - 1) / NTHREADS, NTHREADS, 0, stream>>>(ws_amin);
    vq_norme_kernel<<<NUM_EMB, NTHREADS, 0, stream>>>(emb, ws_normE);

    // M fastest (x), N slow (y): B strip reused across all M blocks, A stays in L2
    dim3 grid(N_TOK / TM, (NUM_EMB + TN - 1) / TN);   // 16 x 32
    vq_scores_argmin_kernel<<<grid, NTHREADS, 0, stream>>>(z, emb, ws_normE, ws_amin);

    vq_gather_kernel<<<N_TOK, NTHREADS, 0, stream>>>(z, emb, ws_amin, out, ws_partial);
    vq_loss_kernel<<<1, NTHREADS, 0, stream>>>(ws_partial, out);
}
